// KronQRInjectedLinear_SVD3_84447646974062
// MI455X (gfx1250) — matrix operand order repaired and checked by reading, not compiled, not run on hardware
//
#include <hip/hip_runtime.h>
#include <hip/hip_bf16.h>

// ---------------------------------------------------------------------------
// KronQRInjectedLinear_SVD3 for MI455X (gfx1250, wave32, WMMA)
//
//   S_eff = relu(S + delta)
//   Us    = U * S_eff (column scale)            -> bf16           [prep_us]
//   Vt    = V^T                                 -> bf16           [transpose_v]
//   Y     = x @ kron(Q1,Q2,Q3)  (3 mode prods)  -> bf16           [kron_kernel]
//   W     = Us @ V   == NT-gemm(Us, Vt)         -> bf16           [gemm_nt]
//   out   = Y @ W^T  == NT-gemm(Y,  W )         -> f32            [gemm_nt]
//
// Workspace layout (160 MB): Us 32MB | Vt 32MB | W 32MB | Y 64MB
// GEMM: block tile 256x128x32, 8 waves (4x2), wave tile 64x64 = 16 WMMA/k-step.
// Global->LDS uses GLOBAL_LOAD_ASYNC_TO_LDS_B128 (ASYNCcnt) when the toolchain
// exposes the builtins, else register-staged loads (both double-buffered).
// ---------------------------------------------------------------------------

typedef __attribute__((ext_vector_type(16))) __bf16 v16bf;
typedef __attribute__((ext_vector_type(8)))  __bf16 v8bf;
typedef __attribute__((ext_vector_type(8)))  float  v8f;

#define IN_F  4096
#define BLK_M 256
#define BLK_N 128
#define LDT   40   // padded LDS K-stride (elements): 80B row stride, bank-spread

#if defined(__has_builtin)
#if __has_builtin(__builtin_amdgcn_global_load_async_to_lds_b128) && \
    __has_builtin(__builtin_amdgcn_s_wait_asynccnt)
#define USE_ASYNC_LDS 1
#endif
#endif
#ifndef USE_ASYNC_LDS
#define USE_ASYNC_LDS 0
#endif

#if USE_ASYNC_LDS
#define AS1 __attribute__((address_space(1)))
#define AS3 __attribute__((address_space(3)))
typedef int b128_t __attribute__((vector_size(4 * sizeof(int))));
__device__ __forceinline__ void async_cp16(const unsigned short* g, unsigned short* l) {
    __builtin_amdgcn_global_load_async_to_lds_b128(
        (AS1 b128_t*)(AS1 void*)g, (AS3 b128_t*)(AS3 void*)l, 0, 0);
}
#endif

__device__ __forceinline__ unsigned short f2bf(float f) {
    unsigned int u = __float_as_uint(f);
    u += 0x7FFFu + ((u >> 16) & 1u);          // round-to-nearest-even
    return (unsigned short)(u >> 16);
}

// ---------------------------------------------------------------- prep_us ---
__global__ __launch_bounds__(256) void prep_us(const float* __restrict__ U,
                                               const float* __restrict__ S,
                                               const float* __restrict__ D,
                                               unsigned short* __restrict__ Us) {
    long long base = ((long long)blockIdx.x * 256 + threadIdx.x) * 4;
    int r = (int)(base & (IN_F - 1));
    float4 u = *(const float4*)(U + base);
    float4 s = *(const float4*)(S + r);
    float4 d = *(const float4*)(D + r);
    float e0 = fmaxf(s.x + d.x, 0.f), e1 = fmaxf(s.y + d.y, 0.f);
    float e2 = fmaxf(s.z + d.z, 0.f), e3 = fmaxf(s.w + d.w, 0.f);
    uint2 o;
    o.x = (unsigned)f2bf(u.x * e0) | ((unsigned)f2bf(u.y * e1) << 16);
    o.y = (unsigned)f2bf(u.z * e2) | ((unsigned)f2bf(u.w * e3) << 16);
    *(uint2*)(Us + base) = o;
}

// ------------------------------------------------------------ transpose_v ---
__global__ __launch_bounds__(256) void transpose_v(const float* __restrict__ V,
                                                   unsigned short* __restrict__ Vt) {
    __shared__ float tile[32][33];
    int tx = threadIdx.x & 31, ty = threadIdx.x >> 5;   // 32 x 8
    long long bx = (long long)blockIdx.x * 32;          // j (col of V)
    long long by = (long long)blockIdx.y * 32;          // r (row of V)
#pragma unroll
    for (int i = 0; i < 4; ++i)
        tile[ty + i * 8][tx] = V[(by + ty + i * 8) * IN_F + bx + tx];
    __syncthreads();
#pragma unroll
    for (int i = 0; i < 4; ++i)
        Vt[(bx + ty + i * 8) * IN_F + by + tx] = f2bf(tile[tx][ty + i * 8]);
}

// ------------------------------------------------------------ kron_kernel ---
// One block per row of x: y[row, i1*256+i2*16+i3] =
//   sum_{j1,j2,j3} x[row, j1*256+j2*16+j3] * Q1[j1,i1]*Q2[j2,i2]*Q3[j3,i3]
__global__ __launch_bounds__(256) void kron_kernel(const float* __restrict__ x,
                                                   const float* __restrict__ Q1,
                                                   const float* __restrict__ Q2,
                                                   const float* __restrict__ Q3,
                                                   unsigned short* __restrict__ y) {
    __shared__ float bufA[IN_F];
    __shared__ float bufB[IN_F];
    __shared__ float q1[256], q2[256], q3[256];
    const int t = threadIdx.x;
    const long long row = blockIdx.x;
    const float* xr = x + row * IN_F;

    q1[t] = Q1[t]; q2[t] = Q2[t]; q3[t] = Q3[t];
#pragma unroll
    for (int i = 0; i < 4; ++i)
        ((float4*)bufA)[t + 256 * i] = ((const float4*)xr)[t + 256 * i];
    __syncthreads();

    float xi[16];
    // mode 3: contract j3 with Q3[j3][i3]
    {
#pragma unroll
        for (int j = 0; j < 16; ++j) xi[j] = bufA[t * 16 + j];
#pragma unroll
        for (int i3 = 0; i3 < 16; ++i3) {
            float s = 0.f;
#pragma unroll
            for (int j = 0; j < 16; ++j) s = fmaf(xi[j], q3[j * 16 + i3], s);
            bufB[t * 16 + i3] = s;
        }
    }
    __syncthreads();
    // mode 2: contract j2 with Q2[j2][i2]
    {
        int j1 = t >> 4, i3 = t & 15;
#pragma unroll
        for (int j = 0; j < 16; ++j) xi[j] = bufB[j1 * 256 + j * 16 + i3];
#pragma unroll
        for (int i2 = 0; i2 < 16; ++i2) {
            float s = 0.f;
#pragma unroll
            for (int j = 0; j < 16; ++j) s = fmaf(xi[j], q2[j * 16 + i2], s);
            bufA[j1 * 256 + i2 * 16 + i3] = s;
        }
    }
    __syncthreads();
    // mode 1: contract j1 with Q1[j1][i1], emit bf16
    {
        int i2 = t >> 4, i3 = t & 15;
#pragma unroll
        for (int j = 0; j < 16; ++j) xi[j] = bufA[j * 256 + i2 * 16 + i3];
#pragma unroll
        for (int i1 = 0; i1 < 16; ++i1) {
            float s = 0.f;
#pragma unroll
            for (int j = 0; j < 16; ++j) s = fmaf(xi[j], q1[j * 16 + i1], s);
            y[row * IN_F + i1 * 256 + i2 * 16 + i3] = f2bf(s);
        }
    }
}

// ---------------------------------------------------------------- gemm_nt ---
// C[M,N] = A[M,K] (row-major bf16) * B[N,K]^T (row-major bf16), f32 accum.
__device__ __forceinline__ void stor(float* p, float v)          { *p = v; }
__device__ __forceinline__ void stor(unsigned short* p, float v) { *p = f2bf(v); }

template <typename OutT>
__global__ __launch_bounds__(256) void gemm_nt(const unsigned short* __restrict__ A,
                                               const unsigned short* __restrict__ B,
                                               OutT* __restrict__ C,
                                               int M, int N, int K) {
    __shared__ unsigned short As[2][BLK_M * LDT];   // 2 x 20 KB
    __shared__ unsigned short Bs[2][BLK_N * LDT];   // 2 x 10 KB

    const int tid  = threadIdx.x;
    const int lane = tid & 31;
    const int wid  = tid >> 5;
    const int wm   = wid >> 1;          // 0..3 -> 64-row slab
    const int wn   = wid & 1;           // 0..1 -> 64-col slab
    const int g    = lane >> 4;         // K-half selector (k = 16g..16g+15, contiguous)
    const int ml   = lane & 15;         // row/col within 16

    const long long rowBase = (long long)blockIdx.y * BLK_M;
    const long long colBase = (long long)blockIdx.x * BLK_N;

    // global->LDS ownership: A row tid (64B), B row tid>>1 half (32B)
    const int brow = tid >> 1;
    const int blk  = (tid & 1) * 16;
    const unsigned short* Aptr = A + (rowBase + tid) * (long long)K;
    const unsigned short* Bptr = B + (colBase + brow) * (long long)K + blk;
    const int aoff = tid * LDT;
    const int boff = brow * LDT + blk;

    v8f acc[4][4];
#pragma unroll
    for (int m = 0; m < 4; ++m)
#pragma unroll
        for (int n = 0; n < 4; ++n) {
            v8f z = {0.f, 0.f, 0.f, 0.f, 0.f, 0.f, 0.f, 0.f};
            acc[m][n] = z;
        }

    const int nk = K >> 5;
    int buf = 0;

#if USE_ASYNC_LDS
    // prime buffer 0 via async DMA (tracked with ASYNCcnt)
    async_cp16(Aptr,      &As[0][aoff]);
    async_cp16(Aptr + 8,  &As[0][aoff + 8]);
    async_cp16(Aptr + 16, &As[0][aoff + 16]);
    async_cp16(Aptr + 24, &As[0][aoff + 24]);
    async_cp16(Bptr,      &Bs[0][boff]);
    async_cp16(Bptr + 8,  &Bs[0][boff + 8]);
    __builtin_amdgcn_s_wait_asynccnt(0);
    __syncthreads();
#else
    {
        uint4 ra0 = *(const uint4*)(Aptr);
        uint4 ra1 = *(const uint4*)(Aptr + 8);
        uint4 ra2 = *(const uint4*)(Aptr + 16);
        uint4 ra3 = *(const uint4*)(Aptr + 24);
        uint4 rb0 = *(const uint4*)(Bptr);
        uint4 rb1 = *(const uint4*)(Bptr + 8);
        *(uint4*)&As[0][aoff]      = ra0;
        *(uint4*)&As[0][aoff + 8]  = ra1;
        *(uint4*)&As[0][aoff + 16] = ra2;
        *(uint4*)&As[0][aoff + 24] = ra3;
        *(uint4*)&Bs[0][boff]      = rb0;
        *(uint4*)&Bs[0][boff + 8]  = rb1;
    }
    __syncthreads();
#endif

    for (int kt = 0; kt < nk; ++kt) {
#if USE_ASYNC_LDS
        if (kt + 1 < nk) {   // async-fetch next tile into the other buffer
            const unsigned short* An = Aptr + (long long)(kt + 1) * 32;
            const unsigned short* Bn = Bptr + (long long)(kt + 1) * 32;
            async_cp16(An,      &As[buf ^ 1][aoff]);
            async_cp16(An + 8,  &As[buf ^ 1][aoff + 8]);
            async_cp16(An + 16, &As[buf ^ 1][aoff + 16]);
            async_cp16(An + 24, &As[buf ^ 1][aoff + 24]);
            async_cp16(Bn,      &Bs[buf ^ 1][boff]);
            async_cp16(Bn + 8,  &Bs[buf ^ 1][boff + 8]);
            if (kt + 2 < nk) {  // -> global_prefetch_b8 (into GL2)
                __builtin_prefetch(Aptr + (long long)(kt + 2) * 32, 0, 3);
                __builtin_prefetch(Bptr + (long long)(kt + 2) * 32, 0, 3);
            }
        }
#else
        uint4 ra0, ra1, ra2, ra3, rb0, rb1;
        if (kt + 1 < nk) {
            const unsigned short* An = Aptr + (long long)(kt + 1) * 32;
            const unsigned short* Bn = Bptr + (long long)(kt + 1) * 32;
            ra0 = *(const uint4*)(An);
            ra1 = *(const uint4*)(An + 8);
            ra2 = *(const uint4*)(An + 16);
            ra3 = *(const uint4*)(An + 24);
            rb0 = *(const uint4*)(Bn);
            rb1 = *(const uint4*)(Bn + 8);
            if (kt + 2 < nk) {
                __builtin_prefetch(Aptr + (long long)(kt + 2) * 32, 0, 3);
                __builtin_prefetch(Bptr + (long long)(kt + 2) * 32, 0, 3);
            }
        }
#endif

        // B fragments: lane L holds col (L%16), K = g*16 .. g*16+15 contiguous
        v16bf bfrag[4];
#pragma unroll
        for (int nt = 0; nt < 4; ++nt) {
            const __bf16* p =
                (const __bf16*)&Bs[buf][(wn * 64 + nt * 16 + ml) * LDT + g * 16];
            v8bf lo = *(const v8bf*)p;
            v8bf hi = *(const v8bf*)(p + 8);
            bfrag[nt] = __builtin_shufflevector(lo, hi, 0, 1, 2, 3, 4, 5, 6, 7,
                                                8, 9, 10, 11, 12, 13, 14, 15);
        }
        // A fragments: lane L holds row (L%16), K = g*16 .. g*16+15 contiguous
#pragma unroll
        for (int mt = 0; mt < 4; ++mt) {
            const __bf16* p =
                (const __bf16*)&As[buf][(wm * 64 + mt * 16 + ml) * LDT + g * 16];
            v8bf lo = *(const v8bf*)p;
            v8bf hi = *(const v8bf*)(p + 8);
            v16bf afrag = __builtin_shufflevector(lo, hi, 0, 1, 2, 3, 4, 5, 6, 7,
                                                  8, 9, 10, 11, 12, 13, 14, 15);
#pragma unroll
            for (int nt = 0; nt < 4; ++nt)
                acc[mt][nt] = __builtin_amdgcn_wmma_f32_16x16x32_bf16(
                    false, afrag, false, bfrag[nt], (short)0, acc[mt][nt],
                    false, false);
        }

#if USE_ASYNC_LDS
        if (kt + 1 < nk) __builtin_amdgcn_s_wait_asynccnt(0);
#else
        if (kt + 1 < nk) {
            *(uint4*)&As[buf ^ 1][aoff]      = ra0;
            *(uint4*)&As[buf ^ 1][aoff + 8]  = ra1;
            *(uint4*)&As[buf ^ 1][aoff + 16] = ra2;
            *(uint4*)&As[buf ^ 1][aoff + 24] = ra3;
            *(uint4*)&Bs[buf ^ 1][boff]      = rb0;
            *(uint4*)&Bs[buf ^ 1][boff + 8]  = rb1;
        }
#endif
        __syncthreads();
        buf ^= 1;
    }

    // C layout: VGPR i of accumulator holds (M = i + 8*g, N = lane%16)
#pragma unroll
    for (int mt = 0; mt < 4; ++mt)
#pragma unroll
        for (int nt = 0; nt < 4; ++nt) {
            long long r0 = rowBase + wm * 64 + mt * 16 + g * 8;
            long long c  = colBase + wn * 64 + nt * 16 + ml;
#pragma unroll
            for (int i = 0; i < 8; ++i)
                stor(&C[(r0 + i) * (long long)N + c], acc[mt][nt][i]);
        }
}

// ----------------------------------------------------------------- launch ---
extern "C" void kernel_launch(void* const* d_in, const int* in_sizes, int n_in,
                              void* d_out, int out_size, void* d_ws, size_t ws_size,
                              hipStream_t stream) {
    (void)in_sizes; (void)n_in; (void)out_size; (void)ws_size;
    const float* x  = (const float*)d_in[0];
    const float* U  = (const float*)d_in[1];
    const float* S  = (const float*)d_in[2];
    const float* V  = (const float*)d_in[3];
    const float* dl = (const float*)d_in[4];
    const float* Q1 = (const float*)d_in[5];
    const float* Q2 = (const float*)d_in[6];
    const float* Q3 = (const float*)d_in[7];
    float* out = (float*)d_out;

    char* ws = (char*)d_ws;
    unsigned short* Us = (unsigned short*)(ws);
    unsigned short* Vt = (unsigned short*)(ws + (32ull << 20));
    unsigned short* W  = (unsigned short*)(ws + (64ull << 20));
    unsigned short* Y  = (unsigned short*)(ws + (96ull << 20));

    // independent prep stages
    prep_us<<<(IN_F * (long long)IN_F) / (4 * 256), 256, 0, stream>>>(U, S, dl, Us);
    transpose_v<<<dim3(IN_F / 32, IN_F / 32), 256, 0, stream>>>(V, Vt);
    kron_kernel<<<8192, 256, 0, stream>>>(x, Q1, Q2, Q3, Y);

    // W = Us @ V   (NT: B = V^T)
    gemm_nt<unsigned short><<<dim3(IN_F / BLK_N, IN_F / BLK_M), 256, 0, stream>>>(
        Us, Vt, W, IN_F, IN_F, IN_F);
    // out = Y @ W^T (NT: B = W row-major)
    gemm_nt<float><<<dim3(IN_F / BLK_N, 8192 / BLK_M), 256, 0, stream>>>(
        Y, W, out, 8192, IN_F, IN_F);
}
